// MultiHeadAttention_13821204759203
// MI455X (gfx1250) — matrix operand order repaired and checked by reading, not compiled, not run on hardware
//
#include <hip/hip_runtime.h>
#include <hip/hip_bf16.h>

typedef _Float16 v4h  __attribute__((ext_vector_type(4)));
typedef _Float16 v8h  __attribute__((ext_vector_type(8)));
typedef _Float16 v16h __attribute__((ext_vector_type(16)));
typedef float    v4f  __attribute__((ext_vector_type(4)));
typedef float    v8f  __attribute__((ext_vector_type(8)));

#define D_MODEL 1024
#define N_HEADS 16
#define D_KH    64
#define SEQ     2048
#define BATCH   2
#define M_TOTAL (BATCH * SEQ)   // 4096 rows for all projection GEMMs

// 1/sqrt(d_k) * log2(e): scores land directly in the exp2 domain.
#define Q_SCALE (0.125f * 1.4426950408889634f)

// ---------------------------------------------------------------------------
// Fragment helpers (wave32 WMMA 16x16x32 f16, f32 accumulate)
// A (16xK=32): lane L holds row M=L&15; K = h*16+0..15 contiguous (h = L>>4)
// B (K=32x16): lanes 0-15 hold K=0..15 for col n=lane&15, lanes 16-31 K=16..31
// C (16x16 f32): vgpr r, lane l -> (M = r + 8*(l>>4), N = l&15)
// ---------------------------------------------------------------------------

static __device__ __forceinline__ v16h join16(v8h lo, v8h hi) {
  return __builtin_shufflevector(lo, hi, 0,1,2,3,4,5,6,7,8,9,10,11,12,13,14,15);
}

static __device__ __forceinline__ v16h load_a16x32(const _Float16* base, int ld,
                                                   int lane, int k0) {
  const int r = lane & 15, h = lane >> 4;
  const _Float16* p = base + (size_t)r * ld + k0 + h * 16;
  v8h lo = *(const v8h*)(p);
  v8h hi = *(const v8h*)(p + 8);
  return join16(lo, hi);
}

// B fragment where memory is laid out "column n is a row": element (k,n) at
// base + n*ld + k. 32 contiguous bytes per lane.
static __device__ __forceinline__ v16h load_b32x16(const _Float16* base, int ld,
                                                   int lane, int k0) {
  const int n = lane & 15, g = lane >> 4;
  const _Float16* p = base + (size_t)n * ld + k0 + g * 16;
  return *(const v16h*)p;
}

static __device__ __forceinline__ v8f wmma_f16(v16h a, v16h b, v8f c) {
  return __builtin_amdgcn_wmma_f32_16x16x32_f16(false, a, false, b,
                                                (short)0, c, false, false);
}

// ---------------------------------------------------------------------------
// fp32 -> fp16 conversion, vectorized (b128 in, b64 out)
// ---------------------------------------------------------------------------
__global__ void __launch_bounds__(256)
cvt_f32_to_f16(const float* __restrict__ src, _Float16* __restrict__ dst, int n4) {
  int i = blockIdx.x * blockDim.x + threadIdx.x;
  const int stride = gridDim.x * blockDim.x;
  for (; i < n4; i += stride) {
    v4f x = *(const v4f*)(src + 4 * (size_t)i);
    v4h y;
#pragma unroll
    for (int j = 0; j < 4; ++j) y[j] = (_Float16)x[j];
    *(v4h*)(dst + 4 * (size_t)i) = y;
  }
}

// ---------------------------------------------------------------------------
// Projection GEMM: C[m,n] = sum_k A[m,k] * W[n,k] + bias[n]
// A: (M_TOTAL x 1024) f16 row-major, W: (1024 x 1024) f16 (out,in) row-major.
// Each wave computes a 32x64 tile; K loop step 32 -> 8 WMMAs / step.
// MODE 0: f16 row-major out, value = (acc + bias) * scale   (Q, K)
// MODE 1: f16 transposed-per-head out (b,h,d,s)             (V)
// MODE 2: f32 row-major out (final output projection)
// ---------------------------------------------------------------------------
template <int MODE>
__global__ void __launch_bounds__(256)
proj_gemm(const _Float16* __restrict__ A, const _Float16* __restrict__ W,
          const float* __restrict__ bias, void* __restrict__ outv, float scale) {
  const int lane = threadIdx.x & 31;
  const int wave = blockIdx.x * (blockDim.x >> 5) + (threadIdx.x >> 5);
  const int mt = wave >> 4;     // M_TOTAL/32 = 128 tiles
  const int nt = wave & 15;     // 1024/64   = 16 tiles
  const int m0 = mt * 32, n0 = nt * 64;

  v8f c[2][4];
#pragma unroll
  for (int i = 0; i < 2; ++i)
#pragma unroll
    for (int j = 0; j < 4; ++j) c[i][j] = v8f{};

  for (int k0 = 0; k0 < D_MODEL; k0 += 32) {
    v16h a0 = load_a16x32(A + (size_t)m0 * D_MODEL, D_MODEL, lane, k0);
    v16h a1 = load_a16x32(A + (size_t)(m0 + 16) * D_MODEL, D_MODEL, lane, k0);
#pragma unroll
    for (int j = 0; j < 4; ++j) {
      v16h b = load_b32x16(W + (size_t)(n0 + j * 16) * D_MODEL, D_MODEL, lane, k0);
      c[0][j] = wmma_f16(a0, b, c[0][j]);
      c[1][j] = wmma_f16(a1, b, c[1][j]);
    }
  }

  const int col = lane & 15, hf = lane >> 4;
#pragma unroll
  for (int j = 0; j < 4; ++j) {
    const int n = n0 + j * 16 + col;
    const float bn = bias[n];
#pragma unroll
    for (int i = 0; i < 2; ++i) {
#pragma unroll
      for (int r = 0; r < 8; ++r) {
        const int m = m0 + i * 16 + r + 8 * hf;
        const float v = (c[i][j][r] + bn) * scale;
        if (MODE == 0) {
          ((_Float16*)outv)[(size_t)m * D_MODEL + n] = (_Float16)v;
        } else if (MODE == 1) {
          const int b = m >> 11, s = m & (SEQ - 1);
          const int h = n >> 6, d = n & (D_KH - 1);
          ((_Float16*)outv)[((size_t)((b * N_HEADS + h) * D_KH + d)) * SEQ + s] =
              (_Float16)v;
        } else {
          ((float*)outv)[(size_t)m * D_MODEL + n] = v;
        }
      }
    }
  }
}

// ---------------------------------------------------------------------------
// Flash attention: one wave owns 16 query rows of one (b,h). Streams keys in
// tiles of 64 with online softmax in the exp2 domain (log2e folded into Q).
// Row-sums come from an extra WMMA against an all-ones B matrix (result lands
// identically in all 16 columns -> every lane holds its rows' sums, no
// shuffles). Only the row-max needs a ds_bpermute butterfly, once per 64 keys.
// ---------------------------------------------------------------------------
__global__ void __launch_bounds__(256)
flash_attn(const _Float16* __restrict__ qp, const _Float16* __restrict__ kp,
           const _Float16* __restrict__ vT, _Float16* __restrict__ ctx) {
  __shared__ _Float16 ldsP[8][16][64];   // 2 KB per wave
  const int lane = threadIdx.x & 31;
  const int w = threadIdx.x >> 5;
  const int wave = blockIdx.x * 8 + w;
  const int qtile = wave & 127;          // SEQ/16
  const int bh = wave >> 7;              // 0..31
  const int b = bh >> 4, h = bh & 15;
  const int q0 = qtile * 16;

  const _Float16* qbase = qp + ((size_t)(b * SEQ + q0)) * D_MODEL + h * D_KH;
  const _Float16* kbase = kp + (size_t)b * SEQ * D_MODEL + h * D_KH;
  const _Float16* vbase = vT + (size_t)bh * D_KH * SEQ;

  const v16h aq0 = load_a16x32(qbase, D_MODEL, lane, 0);
  const v16h aq1 = load_a16x32(qbase, D_MODEL, lane, 32);

  v16h ones;
#pragma unroll
  for (int i = 0; i < 16; ++i) ones[i] = (_Float16)1.0f;

  float mrow[8], lrow[8];
#pragma unroll
  for (int r = 0; r < 8; ++r) { mrow[r] = -1e30f; lrow[r] = 0.0f; }
  v8f o[4];
#pragma unroll
  for (int f = 0; f < 4; ++f) o[f] = v8f{};

  const int col = lane & 15, hf = lane >> 4;
  _Float16* lp = &ldsP[w][0][0];

  for (int j = 0; j < SEQ; j += 64) {
    // ---- scores: four 16x16 tiles covering keys j..j+63 ----
    v8f s[4];
#pragma unroll
    for (int t = 0; t < 4; ++t) {
      s[t] = v8f{};
      v16h b0 = load_b32x16(kbase + (size_t)(j + 16 * t) * D_MODEL, D_MODEL, lane, 0);
      s[t] = wmma_f16(aq0, b0, s[t]);
      v16h b1 = load_b32x16(kbase + (size_t)(j + 16 * t) * D_MODEL, D_MODEL, lane, 32);
      s[t] = wmma_f16(aq1, b1, s[t]);
    }

    // ---- row max: combine 4 tiles in-lane, one butterfly per 64 keys ----
    float alpha[8];
#pragma unroll
    for (int r = 0; r < 8; ++r) {
      float mx = fmaxf(fmaxf(s[0][r], s[1][r]), fmaxf(s[2][r], s[3][r]));
#pragma unroll
      for (int off = 8; off > 0; off >>= 1)
        mx = fmaxf(mx, __shfl_xor(mx, off, 32));
      const float mn = fmaxf(mrow[r], mx);
      alpha[r] = __builtin_amdgcn_exp2f(mrow[r] - mn);
      mrow[r] = mn;
#pragma unroll
      for (int t = 0; t < 4; ++t)
        s[t][r] = __builtin_amdgcn_exp2f(s[t][r] - mn);
    }

    // ---- P (C-layout) -> LDS row-major f16 16x64 ----
#pragma unroll
    for (int r = 0; r < 8; ++r) {
      const int mr = r + 8 * hf;
#pragma unroll
      for (int t = 0; t < 4; ++t)
        lp[mr * 64 + 16 * t + col] = (_Float16)s[t][r];
    }
    // LDS ops from one wave are in order; compiler inserts s_wait_dscnt.
    const v16h ap0 = load_a16x32(lp, 64, lane, 0);
    const v16h ap1 = load_a16x32(lp, 64, lane, 32);

    // ---- row sums via WMMA against ones: every lane gets its rows' sums ----
    v8f rsum = v8f{};
    rsum = wmma_f16(ap0, ones, rsum);
    rsum = wmma_f16(ap1, ones, rsum);
#pragma unroll
    for (int r = 0; r < 8; ++r) lrow[r] = lrow[r] * alpha[r] + rsum[r];

    // ---- rescale running output, then O += P @ V_tile ----
#pragma unroll
    for (int f = 0; f < 4; ++f)
#pragma unroll
      for (int r = 0; r < 8; ++r) o[f][r] *= alpha[r];

#pragma unroll
    for (int f = 0; f < 4; ++f) {
      v16h bv0 = load_b32x16(vbase + (size_t)(f * 16) * SEQ, SEQ, lane, j);
      o[f] = wmma_f16(ap0, bv0, o[f]);
      v16h bv1 = load_b32x16(vbase + (size_t)(f * 16) * SEQ, SEQ, lane, j + 32);
      o[f] = wmma_f16(ap1, bv1, o[f]);
    }
  }

  // ---- normalize and store ctx in (b, s, h*64+d) f16 row-major ----
#pragma unroll
  for (int r = 0; r < 8; ++r) lrow[r] = 1.0f / lrow[r];
#pragma unroll
  for (int f = 0; f < 4; ++f)
#pragma unroll
    for (int r = 0; r < 8; ++r) {
      const int m = q0 + r + 8 * hf;
      const int n = h * D_KH + f * 16 + col;
      ctx[((size_t)(b * SEQ + m)) * D_MODEL + n] = (_Float16)(o[f][r] * lrow[r]);
    }
}

// ---------------------------------------------------------------------------
// Host-side orchestration
// ---------------------------------------------------------------------------
extern "C" void kernel_launch(void* const* d_in, const int* in_sizes, int n_in,
                              void* d_out, int out_size, void* d_ws, size_t ws_size,
                              hipStream_t stream) {
  (void)in_sizes; (void)n_in; (void)out_size; (void)ws_size;

  const float* q  = (const float*)d_in[0];
  const float* k  = (const float*)d_in[1];
  const float* v  = (const float*)d_in[2];
  const float* wq = (const float*)d_in[3];
  const float* wk = (const float*)d_in[4];
  const float* wv = (const float*)d_in[5];
  const float* wo = (const float*)d_in[6];
  const float* bq = (const float*)d_in[7];
  const float* bk = (const float*)d_in[8];
  const float* bv = (const float*)d_in[9];
  const float* bo = (const float*)d_in[10];

  const size_t ACT = (size_t)M_TOTAL * D_MODEL;  // 4M elements (8 MiB f16)
  const size_t WEL = (size_t)D_MODEL * D_MODEL;  // 1M elements (2 MiB f16)

  _Float16* ws    = (_Float16*)d_ws;             // 64 MiB total
  _Float16* xq    = ws;
  _Float16* xk    = xq + ACT;
  _Float16* xv    = xk + ACT;
  _Float16* hwq   = xv + ACT;
  _Float16* hwk   = hwq + WEL;
  _Float16* hwv   = hwk + WEL;
  _Float16* hwo   = hwv + WEL;
  _Float16* qproj = hwo + WEL;                   // (b,s,h,d) f16, exp2-scaled
  _Float16* kproj = qproj + ACT;                 // (b,s,h,d) f16
  _Float16* vprjT = kproj + ACT;                 // (b,h,d,s) f16
  _Float16* ctx   = vprjT + ACT;                 // (b,s,1024) f16

  const dim3 blk(256);

  cvt_f32_to_f16<<<512, blk, 0, stream>>>(q, xq, (int)(ACT / 4));
  cvt_f32_to_f16<<<512, blk, 0, stream>>>(k, xk, (int)(ACT / 4));
  cvt_f32_to_f16<<<512, blk, 0, stream>>>(v, xv, (int)(ACT / 4));
  cvt_f32_to_f16<<<256, blk, 0, stream>>>(wq, hwq, (int)(WEL / 4));
  cvt_f32_to_f16<<<256, blk, 0, stream>>>(wk, hwk, (int)(WEL / 4));
  cvt_f32_to_f16<<<256, blk, 0, stream>>>(wv, hwv, (int)(WEL / 4));
  cvt_f32_to_f16<<<256, blk, 0, stream>>>(wo, hwo, (int)(WEL / 4));

  // 2048 waves per projection GEMM -> 256 blocks of 8 waves
  proj_gemm<0><<<256, blk, 0, stream>>>(xq, hwq, bq, qproj, Q_SCALE);
  proj_gemm<0><<<256, blk, 0, stream>>>(xk, hwk, bk, kproj, 1.0f);
  proj_gemm<1><<<256, blk, 0, stream>>>(xv, hwv, bv, vprjT, 1.0f);

  // 4096 waves (B*H*S/16) -> 512 blocks of 8 waves
  flash_attn<<<512, blk, 0, stream>>>(qproj, kproj, vprjT, ctx);

  // Output projection, f32 + bias straight into d_out
  proj_gemm<2><<<256, blk, 0, stream>>>(ctx, hwo, bo, d_out, 1.0f);
}
